// Merge_Model_45913200394568
// MI455X (gfx1250) — matrix operand order repaired and checked by reading, not compiled, not run on hardware
//
#include <hip/hip_runtime.h>

// ---------------------------------------------------------------------------
// Merge_Model on MI455X (gfx1250, wave32, WMMA).
//   - All 11 D=300 GEMMs via v_wmma_f32_16x16x32_bf16 (K padded to 320,
//     N padded to 304 = 19 tiles of 16).
//   - Edge aggregation via hardware fp32 global atomics (bandwidth-bound path).
//   - mean over doc length commuted before W_dense (linear), saving ~93 GFLOP.
// ---------------------------------------------------------------------------

#define V_NODES 30000
#define DIM     300
#define KPAD    320
#define NPAD    304
#define NTILES  19          // ceil(300/16)
#define NEDGE   960000
#define NHEADS  30
#define DKH     10
#define NDOCS   8192
#define DOCLEN  64
#define NCLS    20

typedef __attribute__((ext_vector_type(16))) __bf16 bf16x16;
typedef __attribute__((ext_vector_type(8)))  __bf16 bf16x8;
typedef __attribute__((ext_vector_type(8)))  float  f32x8;

__device__ __forceinline__ unsigned short f32_to_bf16(float f) {
  unsigned int u = __float_as_uint(f);
  u += 0x7FFFu + ((u >> 16) & 1u);          // round-to-nearest-even
  return (unsigned short)(u >> 16);
}

// ---------------- degree / normalization ----------------
__global__ void k_deg(const int* __restrict__ src, const int* __restrict__ dst,
                      float* ns, float* nd, int ne) {
  int i = blockIdx.x * blockDim.x + threadIdx.x;
  if (i < ne) {
    unsafeAtomicAdd(&ns[src[i]], 1.0f);
    unsafeAtomicAdd(&nd[dst[i]], 1.0f);
  }
}

__global__ void k_rsqrt(float* ns, float* nd, int n) {
  int i = blockIdx.x * blockDim.x + threadIdx.x;
  if (i < n) {
    ns[i] = rsqrtf(fmaxf(ns[i], 1.0f));
    nd[i] = rsqrtf(fmaxf(nd[i], 1.0f));
  }
}

// ---------------- fp32 -> bf16 staging (padded) ----------------
__global__ void k_conv_a(const float* __restrict__ s, unsigned short* __restrict__ d, int m) {
  int i = blockIdx.x * blockDim.x + threadIdx.x;
  int total = m * KPAD;
  if (i < total) {
    int r = i / KPAD, k = i - r * KPAD;
    float v = (k < DIM) ? s[(size_t)r * DIM + k] : 0.0f;
    d[i] = f32_to_bf16(v);
  }
}

// W [300x300] row-major -> Wt bf16 [304][320]  (Wt[n][k] = W[k][n], zero pad)
__global__ void k_conv_w(const float* __restrict__ w, unsigned short* __restrict__ d) {
  int i = blockIdx.x * blockDim.x + threadIdx.x;
  if (i < NPAD * KPAD) {
    int n = i / KPAD, k = i - n * KPAD;
    float v = (n < DIM && k < DIM) ? w[(size_t)k * DIM + n] : 0.0f;
    d[i] = f32_to_bf16(v);
  }
}

// ---------------- WMMA GEMM: C[M,300] = A[M,320]bf16 @ Wt^T + bias ----------
__global__ void k_gemm(const unsigned short* __restrict__ A,
                       const unsigned short* __restrict__ Bt,
                       float* __restrict__ C, const float* __restrict__ bias, int M) {
  int mtile = blockIdx.x * 4 + (threadIdx.x >> 5);  // 4 waves / block
  if (mtile * 16 >= M) return;                      // wave-uniform exit
  int ntile = blockIdx.y;
  int lane = threadIdx.x & 31;
  int r  = lane & 15;
  int kg = lane >> 4;

  const __bf16* Arow = (const __bf16*)A + (size_t)(mtile * 16 + r) * KPAD;
  const __bf16* Brow = (const __bf16*)Bt + (size_t)(ntile * 16 + r) * KPAD;

  f32x8 acc = {};
#pragma unroll
  for (int k = 0; k < KPAD; k += 32) {
    bf16x8 a0 = *(const bf16x8*)(Arow + k + kg * 8);
    bf16x8 a1 = *(const bf16x8*)(Arow + k + 16 + kg * 8);
    bf16x8 b0 = *(const bf16x8*)(Brow + k + kg * 8);
    bf16x8 b1 = *(const bf16x8*)(Brow + k + 16 + kg * 8);
    bf16x16 av, bv;
#pragma unroll
    for (int i = 0; i < 8; ++i) { av[i] = a0[i]; av[8 + i] = a1[i];
                                  bv[i] = b0[i]; bv[8 + i] = b1[i]; }
    acc = __builtin_amdgcn_wmma_f32_16x16x32_bf16(false, av, false, bv,
                                                  (short)0, acc, false, false);
  }

  int n = ntile * 16 + r;
  if (n < DIM) {
    float bb = bias ? bias[n] : 0.0f;
    int m0 = mtile * 16 + kg * 8;
#pragma unroll
    for (int i = 0; i < 8; ++i) C[(size_t)(m0 + i) * DIM + n] = acc[i] + bb;
  }
}

// ---------------- edge scatter: agg[dst] += h[src] * ns[src] ----------------
__global__ void k_scatter(const float* __restrict__ h, const float* __restrict__ ns,
                          const int* __restrict__ src, const int* __restrict__ dst,
                          float* __restrict__ agg, int total) {
  int i = blockIdx.x * blockDim.x + threadIdx.x;
  if (i < total) {
    int e = i / DIM, d = i - e * DIM;
    int s = src[e], t = dst[e];
    unsafeAtomicAdd(&agg[(size_t)t * DIM + d], h[(size_t)s * DIM + d] * ns[s]);
  }
}

// layer1 = agg * nd + b
__global__ void k_scale_bias(const float* __restrict__ agg, const float* __restrict__ nd,
                             const float* __restrict__ b, float* __restrict__ out) {
  int i = blockIdx.x * blockDim.x + threadIdx.x;
  if (i < V_NODES * DIM) {
    int v = i / DIM, d = i - v * DIM;
    out[i] = agg[i] * nd[v] + b[d];
  }
}

// cat[(v*3+slot)] = agg * nd + b + emb (residual), interleaved for MHA GEMMs
__global__ void k_final_gcn(const float* __restrict__ agg, const float* __restrict__ nd,
                            const float* __restrict__ b, const float* __restrict__ emb,
                            float* __restrict__ cat, int slot) {
  int i = blockIdx.x * blockDim.x + threadIdx.x;
  if (i < V_NODES * DIM) {
    int v = i / DIM, d = i - v * DIM;
    cat[(size_t)(v * 3 + slot) * DIM + d] = agg[i] * nd[v] + b[d] + emb[i];
  }
}

// ---------------- 30-head attention over 3 positions ----------------
__global__ void k_attn(const float* __restrict__ q, const float* __restrict__ k,
                       const float* __restrict__ v, float* __restrict__ o) {
  int t = blockIdx.x * blockDim.x + threadIdx.x;
  if (t >= V_NODES * NHEADS) return;
  int n = t / NHEADS, h = t - n * NHEADS;
  int base = h * DKH;
  float qq[3][DKH], kk[3][DKH], vv[3][DKH];
#pragma unroll
  for (int s = 0; s < 3; ++s) {
    int row = (n * 3 + s) * DIM + base;
#pragma unroll
    for (int d = 0; d < DKH; ++d) { qq[s][d] = q[row + d]; kk[s][d] = k[row + d]; vv[s][d] = v[row + d]; }
  }
  const float scale = 0.3162277660168379f;  // 1/sqrt(10)
  float att[3][3];
#pragma unroll
  for (int qp = 0; qp < 3; ++qp) {
    float sc[3];
#pragma unroll
    for (int kp = 0; kp < 3; ++kp) {
      float s_ = 0.0f;
#pragma unroll
      for (int d = 0; d < DKH; ++d) s_ += qq[qp][d] * kk[kp][d];
      sc[kp] = s_ * scale;
    }
    float m = fmaxf(sc[0], fmaxf(sc[1], sc[2]));
    float e0 = __expf(sc[0] - m), e1 = __expf(sc[1] - m), e2 = __expf(sc[2] - m);
    float inv = 1.0f / (e0 + e1 + e2);
    att[qp][0] = e0 * inv; att[qp][1] = e1 * inv; att[qp][2] = e2 * inv;
  }
#pragma unroll
  for (int qp = 0; qp < 3; ++qp) {
    int row = (n * 3 + qp) * DIM + base;
#pragma unroll
    for (int d = 0; d < DKH; ++d)
      o[row + d] = att[qp][0] * vv[0][d] + att[qp][1] * vv[1][d] + att[qp][2] * vv[2][d];
  }
}

// mean over the 3 view positions
__global__ void k_mean3(const float* __restrict__ cat, float* __restrict__ out) {
  int i = blockIdx.x * blockDim.x + threadIdx.x;
  if (i < V_NODES * DIM) {
    int v = i / DIM, d = i - v * DIM;
    out[i] = (cat[(size_t)(3 * v) * DIM + d] + cat[(size_t)(3 * v + 1) * DIM + d] +
              cat[(size_t)(3 * v + 2) * DIM + d]) * (1.0f / 3.0f);
  }
}

// per-doc mean of gathered embeddings (index == V -> zero row)
__global__ void k_docmean(const int* __restrict__ adj, const float* __restrict__ table,
                          float* __restrict__ out) {
  int i = blockIdx.x * blockDim.x + threadIdx.x;
  if (i >= NDOCS * DIM) return;
  int doc = i / DIM, d = i - doc * DIM;
  const int* row = adj + doc * DOCLEN;
  float s = 0.0f;
  for (int l = 0; l < DOCLEN; ++l) {
    int ix = row[l];
    if (ix < V_NODES) s += table[(size_t)ix * DIM + d];
  }
  out[i] = s * (1.0f / DOCLEN);
}

// final FC: [8192,300] @ [300,20] + b
__global__ void k_fc(const float* __restrict__ x, const float* __restrict__ w,
                     const float* __restrict__ b, float* __restrict__ out) {
  int i = blockIdx.x * blockDim.x + threadIdx.x;
  if (i >= NDOCS * NCLS) return;
  int doc = i / NCLS, c = i - doc * NCLS;
  float s = b[c];
  for (int d = 0; d < DIM; ++d) s += x[(size_t)doc * DIM + d] * w[d * NCLS + c];
  out[i] = s;
}

// ---------------------------------------------------------------------------
extern "C" void kernel_launch(void* const* d_in, const int* in_sizes, int n_in,
                              void* d_out, int out_size, void* d_ws, size_t ws_size,
                              hipStream_t stream) {
  (void)in_sizes; (void)n_in; (void)out_size; (void)ws_size;

  const int* adj = (const int*)d_in[0];
  const int* srcs[3] = {(const int*)d_in[1], (const int*)d_in[3], (const int*)d_in[5]};
  const int* dsts[3] = {(const int*)d_in[2], (const int*)d_in[4], (const int*)d_in[6]};
  const float* emb = (const float*)d_in[8];
  const float* W1s[3] = {(const float*)d_in[9],  (const float*)d_in[13], (const float*)d_in[17]};
  const float* b1s[3] = {(const float*)d_in[10], (const float*)d_in[14], (const float*)d_in[18]};
  const float* W2s[3] = {(const float*)d_in[11], (const float*)d_in[15], (const float*)d_in[19]};
  const float* b2s[3] = {(const float*)d_in[12], (const float*)d_in[16], (const float*)d_in[20]};
  const float* Wq = (const float*)d_in[21], *bq = (const float*)d_in[22];
  const float* Wk = (const float*)d_in[23], *bk = (const float*)d_in[24];
  const float* Wv = (const float*)d_in[25], *bv = (const float*)d_in[26];
  const float* Wo = (const float*)d_in[27], *bo = (const float*)d_in[28];
  const float* Wd = (const float*)d_in[29], *bd = (const float*)d_in[30];
  const float* Wfc = (const float*)d_in[31], *bfc = (const float*)d_in[32];
  float* out = (float*)d_out;

  // ---- workspace carve-out ----
  size_t off = 0;
  char* base = (char*)d_ws;
  auto alloc = [&](size_t bytes) { size_t o = off; off += (bytes + 255) & ~(size_t)255; return o; };
  float* ns[3]; float* nd[3];
  for (int g = 0; g < 3; ++g) {
    ns[g] = (float*)(base + alloc((size_t)V_NODES * 4));
    nd[g] = (float*)(base + alloc((size_t)V_NODES * 4));
  }
  unsigned short* Abf = (unsigned short*)(base + alloc((size_t)3 * V_NODES * KPAD * 2));
  unsigned short* Wt  = (unsigned short*)(base + alloc((size_t)NPAD * KPAD * 2));
  float* tmp1  = (float*)(base + alloc((size_t)V_NODES * DIM * 4));
  float* tmp2  = (float*)(base + alloc((size_t)V_NODES * DIM * 4));
  float* cat   = (float*)(base + alloc((size_t)3 * V_NODES * DIM * 4));
  float* qb    = (float*)(base + alloc((size_t)3 * V_NODES * DIM * 4));
  float* kbuf  = (float*)(base + alloc((size_t)3 * V_NODES * DIM * 4));
  float* vbuf  = (float*)(base + alloc((size_t)3 * V_NODES * DIM * 4));
  float* obuf  = (float*)(base + alloc((size_t)3 * V_NODES * DIM * 4));
  float* dense = (float*)(base + alloc((size_t)NDOCS * DIM * 4));

  const int TB = 256;
  auto CONVA = [&](const float* X, int M) {
    int total = M * KPAD;
    k_conv_a<<<(total + TB - 1) / TB, TB, 0, stream>>>(X, Abf, M);
  };
  auto GEMM = [&](const float* Wmat, const float* bias, float* Cout, int M) {
    k_conv_w<<<(NPAD * KPAD + TB - 1) / TB, TB, 0, stream>>>(Wmat, Wt);
    dim3 grd((M / 16 + 3) / 4, NTILES);
    k_gemm<<<grd, dim3(128), 0, stream>>>(Abf, Wt, Cout, bias, M);
  };

  const int vdTotal = V_NODES * DIM;
  const int edTotal = NEDGE * DIM;  // 288,000,000 < 2^31

  // ---- three 2-layer GCN branches ----
  for (int g = 0; g < 3; ++g) {
    hipMemsetAsync(ns[g], 0, (size_t)V_NODES * 4, stream);
    hipMemsetAsync(nd[g], 0, (size_t)V_NODES * 4, stream);
    k_deg<<<(NEDGE + TB - 1) / TB, TB, 0, stream>>>(srcs[g], dsts[g], ns[g], nd[g], NEDGE);
    k_rsqrt<<<(V_NODES + TB - 1) / TB, TB, 0, stream>>>(ns[g], nd[g], V_NODES);

    // layer 1: h = emb @ W1 ; agg ; *nd + b1
    CONVA(emb, V_NODES);
    GEMM(W1s[g], nullptr, tmp1, V_NODES);
    hipMemsetAsync(tmp2, 0, (size_t)vdTotal * 4, stream);
    k_scatter<<<(edTotal + TB - 1) / TB, TB, 0, stream>>>(tmp1, ns[g], srcs[g], dsts[g], tmp2, edTotal);
    k_scale_bias<<<(vdTotal + TB - 1) / TB, TB, 0, stream>>>(tmp2, nd[g], b1s[g], tmp1);

    // layer 2: h2 = l1 @ W2 ; agg ; *nd + b2 + emb  -> cat[:, g, :]
    CONVA(tmp1, V_NODES);
    GEMM(W2s[g], nullptr, tmp2, V_NODES);
    hipMemsetAsync(tmp1, 0, (size_t)vdTotal * 4, stream);
    k_scatter<<<(edTotal + TB - 1) / TB, TB, 0, stream>>>(tmp2, ns[g], srcs[g], dsts[g], tmp1, edTotal);
    k_final_gcn<<<(vdTotal + TB - 1) / TB, TB, 0, stream>>>(tmp1, nd[g], b2s[g], emb, cat, g);
  }

  // ---- MHA pool over the 3 views ----
  const int M3 = 3 * V_NODES;  // 90000 rows
  CONVA(cat, M3);
  GEMM(Wq, bq, qb, M3);
  GEMM(Wk, bk, kbuf, M3);
  GEMM(Wv, bv, vbuf, M3);
  k_attn<<<(V_NODES * NHEADS + TB - 1) / TB, TB, 0, stream>>>(qb, kbuf, vbuf, obuf);
  CONVA(obuf, M3);
  GEMM(Wo, bo, cat, M3);                      // cat now holds o @ Wo + bo
  k_mean3<<<(vdTotal + TB - 1) / TB, TB, 0, stream>>>(cat, tmp1);  // doc_embedding [V,300]

  // ---- document head (mean commuted before W_dense) ----
  k_docmean<<<(NDOCS * DIM + TB - 1) / TB, TB, 0, stream>>>(adj, tmp1, tmp2);
  CONVA(tmp2, NDOCS);
  GEMM(Wd, bd, dense, NDOCS);
  k_fc<<<(NDOCS * NCLS + TB - 1) / TB, TB, 0, stream>>>(dense, Wfc, bfc, out);
}
